// KANFeedForward_41214506172898
// MI455X (gfx1250) — matrix operand order repaired and checked by reading, not compiled, not run on hardware
//
#include <hip/hip_runtime.h>
#include <hip/hip_bf16.h>
#include <stdint.h>

// KAN feedforward as two fused basis+GEMM passes on gfx1250 WMMA.
// Weights staged to LDS via the Tensor Data Mover (tensor_load_to_lds, TENSORcnt),
// basis computed on the fly into LDS, double-buffered, v_wmma_f32_16x16x32_f16 core.

typedef _Float16 h4  __attribute__((ext_vector_type(4)));
typedef _Float16 h8  __attribute__((ext_vector_type(8)));
typedef _Float16 h16 __attribute__((ext_vector_type(16)));
typedef float    f8  __attribute__((ext_vector_type(8)));
typedef uint32_t u32x4 __attribute__((ext_vector_type(4)));
typedef uint32_t u32x8 __attribute__((ext_vector_type(8)));

#define TILE     128
#define KB       64
#define PITCH    72     // halves per LDS row: 64 + 8 pad -> 144B stride, conflict-free b128 frag loads
#define NTHREADS 256

// ---------------- fp32 -> fp16 weight conversion (x4 vectorized) ----------------
__global__ void cvt_f32_f16(const float* __restrict__ in, _Float16* __restrict__ out, int n4) {
  int i = blockIdx.x * blockDim.x + threadIdx.x;
  if (i < n4) {
    float4 v = ((const float4*)in)[i];
    h4 o;
    o[0] = (_Float16)v.x; o[1] = (_Float16)v.y;
    o[2] = (_Float16)v.z; o[3] = (_Float16)v.w;
    *(h4*)(out + (size_t)i * 4) = o;
  }
}

// ---------------- x (rows x cols) -> xT (cols x rows), fp32 ----------------
__global__ void transpose_f32(const float* __restrict__ in, float* __restrict__ out,
                              int rows, int cols) {
  int idx = blockIdx.x * blockDim.x + threadIdx.x;
  if (idx < rows * cols) {
    int r = idx / cols, c = idx - r * cols;     // coalesced read along c
    out[(size_t)c * rows + r] = in[idx];
  }
}

// ---------------- fused basis + GEMM ----------------
// BASIS_IS_A=false: rows=o (weights=A), cols=n (basis=B)   [layer 1, writes hT]
// BASIS_IS_A=true : rows=n (basis=A),   cols=o (weights=B) [layer 2, writes out]
template<bool BASIS_IS_A>
__global__ void __launch_bounds__(NTHREADS)
kan_gemm(const _Float16* __restrict__ W, const float* __restrict__ Xsrc,
         float* __restrict__ Out, int K, int outPitch, int xPitch) {
  __shared__ __align__(16) _Float16 Wls[2][TILE * PITCH];   // weight tiles (TDM-filled)
  __shared__ __align__(16) _Float16 Sls[2][TILE * PITCH];   // basis tiles

  const int tid  = threadIdx.x;
  const int lane = tid & 31;
  const int wave = tid >> 5;
  const int wr   = wave & 3;     // 4 row-groups of 32
  const int wc   = wave >> 2;    // 2 col-groups of 64
  const int llo  = lane & 15;
  const int kSel = lane >> 4;

  const int rowBase = blockIdx.y * TILE;
  const int colBase = blockIdx.x * TILE;
  const int nBase   = BASIS_IS_A ? rowBase : colBase;
  const int oBase   = BASIS_IS_A ? colBase : rowBase;

  f8 acc[2][4];
#pragma unroll
  for (int i = 0; i < 2; ++i)
#pragma unroll
    for (int j = 0; j < 4; ++j)
#pragma unroll
      for (int e = 0; e < 8; ++e) acc[i][j][e] = 0.0f;

  const float inv_h = 13.0f / 8.0f;   // 1/h, h = 8/13

  // ---- TDM: DMA one 128 x 64-half weight tile into LDS (wave 0 only) ----
  auto stageW = [&](int buf, int kk0) {
    if (wave == 0) {
      uint64_t ga = (uint64_t)(uintptr_t)(W + (size_t)oBase * K + kk0);
      uint32_t la = (uint32_t)(uintptr_t)(&Wls[buf][0]);
      u32x4 g0;
      g0[0] = 1u;                                          // count=1, user descriptor
      g0[1] = la;                                          // lds_addr
      g0[2] = (uint32_t)ga;                                // global_addr[31:0]
      g0[3] = (uint32_t)((ga >> 32) & 0x1FFFFFFu)          // global_addr[56:32]
              | (2u << 30);                                // type = 2 ("image")
      u32x8 g1;
      g1[0] = (1u << 16)                                   // data_size = 2B
            | (1u << 20)                                   // pad_enable
            | (4u << 22)                                   // pad_interval: 32 DWORDs (128B row)
            | (3u << 25);                                  // pad_amount: 4 DWORDs (16B) -> pitch 144B
      g1[1] = 64u  << 16;                                  // tensor_dim0 = 64 (halves)
      g1[2] = 128u << 16;                                  // tensor_dim1 = 128
      g1[3] = 64u  << 16;                                  // tile_dim0 = 64
      g1[4] = 128u;                                        // tile_dim1 = 128, tile_dim2 = 0
      g1[5] = (uint32_t)K;                                 // tensor_dim0_stride (row stride, halves)
      g1[6] = 0u;
      g1[7] = 0u;
      asm volatile("tensor_load_to_lds %0, %1" :: "s"(g0), "s"(g1) : "memory");
    }
  };

  // ---- basis tile: 128 n x 4 d, cubic B-spline closed form ----
  auto stageS = [&](int buf, int kk0) {
#pragma unroll
    for (int it = 0; it < 2; ++it) {
      int idx = it * NTHREADS + tid;            // 0..511
      int nl  = idx & (TILE - 1);
      int dj  = idx >> 7;                       // 0..3
      int d   = (kk0 >> 4) + dj;
      float xv = Xsrc[(size_t)d * xPitch + nBase + nl];
      float u  = fmaf(xv, inv_h, 9.5f);         // (x+4)/h + 3
      int   ci = (int)floorf(u);
      ci = ci < 3 ? 3 : (ci > 15 ? 15 : ci);
      float t = u - (float)ci;
      t = t < 0.0f ? 0.0f : (t > 1.0f ? 1.0f : t);
      float s  = 1.0f - t;
      float t2 = t * t, t3 = t2 * t;
      float b0 = s * s * s * (1.0f / 6.0f);
      float b3 = t3 * (1.0f / 6.0f);
      float b1 = (3.0f * t3 - 6.0f * t2 + 4.0f) * (1.0f / 6.0f);
      float b2 = 1.0f - b0 - b1 - b3;           // partition of unity
      _Float16* seg = &Sls[buf][nl * PITCH + dj * 16];
      h8 z;
#pragma unroll
      for (int q = 0; q < 8; ++q) z[q] = (_Float16)0.0f;
      *(h8*)seg       = z;
      *(h8*)(seg + 8) = z;
      seg[ci - 3] = (_Float16)b0;
      seg[ci - 2] = (_Float16)b1;
      seg[ci - 1] = (_Float16)b2;
      seg[ci    ] = (_Float16)b3;
    }
  };

  // ---- prologue: fill buffer 0 ----
  stageW(0, 0);
  stageS(0, 0);
  if (wave == 0) __builtin_amdgcn_s_wait_tensorcnt(0);
  __syncthreads();

  const int nch = K / KB;
  for (int chunk = 0; chunk < nch; ++chunk) {
    const int cur = chunk & 1;
    if (chunk + 1 < nch) {            // stage next buffer while computing current
      stageW(cur ^ 1, (chunk + 1) * KB);
      stageS(cur ^ 1, (chunk + 1) * KB);
    }

    const _Float16* Atile;
    const _Float16* Btile;
    if constexpr (BASIS_IS_A) { Atile = &Sls[cur][0]; Btile = &Wls[cur][0]; }
    else                      { Atile = &Wls[cur][0]; Btile = &Sls[cur][0]; }

#pragma unroll
    for (int kb = 0; kb < KB; kb += 32) {
      h16 afrag[2];
#pragma unroll
      for (int rt = 0; rt < 2; ++rt) {
        int row = wr * 32 + rt * 16 + llo;
        const _Float16* p = Atile + row * PITCH + kb + kSel * 8;
        h8 lo = *(const h8*)p;
        h8 hi = *(const h8*)(p + 16);
        afrag[rt] = __builtin_shufflevector(lo, hi, 0,1,2,3,4,5,6,7,8,9,10,11,12,13,14,15);
      }
      h16 bfrag[4];
#pragma unroll
      for (int ct = 0; ct < 4; ++ct) {
        int row = wc * 64 + ct * 16 + llo;
        const _Float16* p = Btile + row * PITCH + kb + kSel * 8;
        h8 lo = *(const h8*)p;
        h8 hi = *(const h8*)(p + 16);
        bfrag[ct] = __builtin_shufflevector(lo, hi, 0,1,2,3,4,5,6,7,8,9,10,11,12,13,14,15);
      }
      // ct-outer / rt-inner: both A-fragments live from the first WMMA,
      // avoids register-reuse stall on the A reload.
#pragma unroll
      for (int ct = 0; ct < 4; ++ct)
#pragma unroll
        for (int rt = 0; rt < 2; ++rt)
          acc[rt][ct] = __builtin_amdgcn_wmma_f32_16x16x32_f16(
              false, afrag[rt], false, bfrag[ct], (short)0, acc[rt][ct], false, false);
    }

    if (wave == 0) __builtin_amdgcn_s_wait_tensorcnt(0);   // next tile landed in LDS
    __syncthreads();
  }

  // ---- epilogue: fp32 stores, 32-bit addressing, contiguous 64B per half-wave ----
  float* outp = Out + (uint32_t)(rowBase + wr * 32 + kSel * 8) * (uint32_t)outPitch
                    + (uint32_t)(colBase + wc * 64 + llo);
#pragma unroll
  for (int rt = 0; rt < 2; ++rt) {
#pragma unroll
    for (int ct = 0; ct < 4; ++ct) {
      uint32_t off = (uint32_t)(rt * 16) * (uint32_t)outPitch + (uint32_t)(ct * 16);
#pragma unroll
      for (int r = 0; r < 8; ++r)
        outp[off + (uint32_t)r * (uint32_t)outPitch] = acc[rt][ct][r];
    }
  }
}

extern "C" void kernel_launch(void* const* d_in, const int* in_sizes, int n_in,
                              void* d_out, int out_size, void* d_ws, size_t ws_size,
                              hipStream_t stream) {
  const float* x  = (const float*)d_in[0];   // (2,2048,768) -> 4096 x 768
  const float* w1 = (const float*)d_in[1];   // (2048, 768, 16)
  const float* w2 = (const float*)d_in[2];   // (768, 2048, 16)
  float* out = (float*)d_out;                // 4096 x 768

  const int N = 4096, D = 768, H = 2048, CP = 16;
  const int K1 = D * CP;   // 12288
  const int K2 = H * CP;   // 32768

  char* ws = (char*)d_ws;
  _Float16* w1h = (_Float16*)ws;                                  // H*K1 halves (48MB)
  _Float16* w2h = (_Float16*)(ws + (size_t)H * K1 * 2);           // D*K2 halves (48MB)
  float*    xT  = (float*)  (ws + (size_t)H * K1 * 2 + (size_t)D * K2 * 2);  // D x N (12MB)
  float*    hT  = (float*)  ((char*)xT + (size_t)D * N * 4);      // H x N (32MB)

  int n1 = H * K1 / 4;
  cvt_f32_f16<<<(n1 + 255) / 256, 256, 0, stream>>>(w1, w1h, n1);
  int n2 = D * K2 / 4;
  cvt_f32_f16<<<(n2 + 255) / 256, 256, 0, stream>>>(w2, w2h, n2);
  int nt = N * D;
  transpose_f32<<<(nt + 255) / 256, 256, 0, stream>>>(x, xT, N, D);

  // Layer 1: rows = o (2048), cols = n (4096); weights = A, basis = B; writes hT[o][n]
  kan_gemm<false><<<dim3(N / TILE, H / TILE), NTHREADS, 0, stream>>>(
      w1h, xT, hT, K1, /*outPitch*/ N, /*xPitch*/ N);
  // Layer 2: rows = n (4096), cols = o (768); basis = A, weights = B; writes out[n][o]
  kan_gemm<true><<<dim3(D / TILE, N / TILE), NTHREADS, 0, stream>>>(
      w2h, hT, out, K2, /*outPitch*/ D, /*xPitch*/ N);
}
